// FusedMoEMethod_24275155157410
// MI455X (gfx1250) — matrix operand order repaired and checked by reading, not compiled, not run on hardware
//
#include <hip/hip_runtime.h>

// Fused MoE (SiLU-gated, top-2 of 8 experts), fp32 in/out, bf16 WMMA compute.
// T=8192, H=2048, I=5632, E=8. Memory-bound (>=1.1GB weight stream @ 23.3TB/s),
// so we compute densely per expert (w_e==0 rows contribute exact zeros) and
// convert fp32->bf16 with native v_cvt_pk_bf16_f32 while staging tiles to LDS.
// Workspace layout: [w_all: T*E f32 = 256KB][act: T*I bf16 = 92.3MB] (~93MB total).

#define TT 8192
#define HH 2048
#define II 5632
#define EE 8

#define BM 64
#define BN 64
#define BK 64
#define LDSS 80   // LDS row stride in bf16 elements (160B: 16B-aligned, staggers banks)

typedef __bf16 bf16_t;
typedef __bf16 bf16x4 __attribute__((ext_vector_type(4)));
typedef __bf16 bf16x8 __attribute__((ext_vector_type(8)));
typedef __bf16 v16bf  __attribute__((ext_vector_type(16)));
typedef float  v8f    __attribute__((ext_vector_type(8)));

__device__ __forceinline__ void sched_fence() {
#if __has_builtin(__builtin_amdgcn_sched_barrier)
  __builtin_amdgcn_sched_barrier(0);   // keep frag-load burst and WMMA burst separated
#endif
}

// Native fp32 -> bf16 (RNE) converts; lowers to v_cvt_pk_bf16_f32.
__device__ __forceinline__ bf16x4 f4_to_bf4(float4 v) {
  bf16x4 r;
  r.x = (__bf16)v.x; r.y = (__bf16)v.y;
  r.z = (__bf16)v.z; r.w = (__bf16)v.w;
  return r;
}

// A fragment (16x32 bf16): half h of wave holds K = h*16+[0..15]
__device__ __forceinline__ v16bf ld_fragA(const bf16_t* p) {
  bf16x8 lo = *(const bf16x8*)(p);        // ds_load_b128
  bf16x8 hi = *(const bf16x8*)(p + 8);    // ds_load_b128
  return __builtin_shufflevector(lo, hi, 0,1,2,3,4,5,6,7,8,9,10,11,12,13,14,15);
}
// B fragment (32x16 bf16): K rows striped across lanes per VGPR; half h holds K = h*16+[0..15]
__device__ __forceinline__ v16bf ld_fragB(const bf16_t* p) {
  bf16x8 lo = *(const bf16x8*)(p);
  bf16x8 hi = *(const bf16x8*)(p + 8);
  return __builtin_shufflevector(lo, hi, 0,1,2,3,4,5,6,7,8,9,10,11,12,13,14,15);
}

__device__ __forceinline__ v8f wmma_bf16(v16bf a, v16bf b, v8f c) {
  return __builtin_amdgcn_wmma_f32_16x16x32_bf16(false, a, false, b, (short)0, c, false, false);
}

// ---------------------------------------------------------------------------
// Router: softmax -> top-2 -> renormalize. Renormalized top-2 weights reduce
// to a 2-way softmax over the two largest logits (global denominator cancels).
// Ties resolve to the lower index, matching lax.top_k.
// ---------------------------------------------------------------------------
__global__ __launch_bounds__(256) void moe_router(const float* __restrict__ logits,
                                                  float* __restrict__ w_all) {
  int t = blockIdx.x * 256 + threadIdx.x;
  if (t >= TT) return;
  float l[EE];
#pragma unroll
  for (int e = 0; e < EE; ++e) l[e] = logits[t * EE + e];
  int i1 = 0;
#pragma unroll
  for (int e = 1; e < EE; ++e) if (l[e] > l[i1]) i1 = e;
  int i2 = (i1 == 0) ? 1 : 0;
#pragma unroll
  for (int e = 0; e < EE; ++e) if (e != i1 && l[e] > l[i2]) i2 = e;
  float e2 = __expf(l[i2] - l[i1]);
  float s = 1.f + e2;
  float w1 = 1.f / s, w2 = e2 / s;
#pragma unroll
  for (int e = 0; e < EE; ++e)
    w_all[t * EE + e] = (e == i1) ? w1 : ((e == i2) ? w2 : 0.f);
}

// ---------------------------------------------------------------------------
// GEMM1: act[t, n] = silu(x.w13g[n]) * (x.w13u[n])  for expert e, bf16 out.
// Per workgroup: 64 token rows x 64 act columns (gate col n and up col n+I
// computed with separate accumulators). 8 waves = 4x2 grid of 16x32 wave tiles.
// ---------------------------------------------------------------------------
__global__ __launch_bounds__(256) void moe_gemm1(const float* __restrict__ x,
                                                 const float* __restrict__ w13,
                                                 bf16_t* __restrict__ act,
                                                 int e) {
  __shared__ __align__(16) bf16_t lA [BM * LDSS];
  __shared__ __align__(16) bf16_t lBg[BN * LDSS];
  __shared__ __align__(16) bf16_t lBu[BN * LDSS];

  const int tid   = threadIdx.x;
  const int m0    = blockIdx.y * BM;
  const int n0    = blockIdx.x * BN;
  const int ldrow = tid >> 4;          // 0..15
  const int ldcol = (tid & 15) << 2;   // 0..60
  const int wv = tid >> 5, lane = tid & 31;
  const int wm = wv >> 1, wn = wv & 1;
  const int nl = lane & 15, hf = lane >> 4;

  const float* xa  = &x  [(size_t)(m0 + ldrow) * HH + ldcol];
  const float* bga = &w13[((size_t)e * 2 * II + n0 + ldrow) * HH + ldcol];
  const float* bua = &w13[((size_t)e * 2 * II + II + n0 + ldrow) * HH + ldcol];

  float4 ra[4], rg[4], ru[4];
#pragma unroll
  for (int p = 0; p < 4; ++p) {
    ra[p] = *(const float4*)(xa  + (size_t)p * 16 * HH);
    rg[p] = *(const float4*)(bga + (size_t)p * 16 * HH);
    ru[p] = *(const float4*)(bua + (size_t)p * 16 * HH);
  }

  const v8f vz = {0.f,0.f,0.f,0.f,0.f,0.f,0.f,0.f};
  v8f accg[2] = {vz, vz};
  v8f accu[2] = {vz, vz};

  for (int kb = 0; kb < HH; kb += BK) {
    __syncthreads();
#pragma unroll
    for (int p = 0; p < 4; ++p) {
      const int r = p * 16 + ldrow;
      *(bf16x4*)&lA [r * LDSS + ldcol] = f4_to_bf4(ra[p]);
      *(bf16x4*)&lBg[r * LDSS + ldcol] = f4_to_bf4(rg[p]);
      *(bf16x4*)&lBu[r * LDSS + ldcol] = f4_to_bf4(ru[p]);
    }
    __syncthreads();
    if (kb + BK < HH) {          // register-prefetch next K stage during compute
#pragma unroll
      for (int p = 0; p < 4; ++p) {
        ra[p] = *(const float4*)(xa  + (size_t)p * 16 * HH + kb + BK);
        rg[p] = *(const float4*)(bga + (size_t)p * 16 * HH + kb + BK);
        ru[p] = *(const float4*)(bua + (size_t)p * 16 * HH + kb + BK);
      }
      if (kb + 2 * BK < HH) {    // streaming hint for the stage after next
        __builtin_prefetch(bga + kb + 2 * BK, 0, 0);
        __builtin_prefetch(bua + kb + 2 * BK, 0, 0);
      }
    }
#pragma unroll
    for (int kk = 0; kk < BK; kk += 32) {
      // Burst-issue all 10 ds_load_b128 of this k-step into live registers,
      // fence the scheduler, then drain with partial s_wait_dscnt per WMMA.
      const int b0 = (wn * 32 + nl) * LDSS + hf * 16 + kk;
      const int b1 = b0 + 16 * LDSS;
      v16bf af = ld_fragA(&lA[(wm * 16 + nl) * LDSS + hf * 16 + kk]);
      v16bf g0 = ld_fragB(&lBg[b0]);
      v16bf u0 = ld_fragB(&lBu[b0]);
      v16bf g1 = ld_fragB(&lBg[b1]);
      v16bf u1 = ld_fragB(&lBu[b1]);
      sched_fence();
      accg[0] = wmma_bf16(af, g0, accg[0]);
      accu[0] = wmma_bf16(af, u0, accu[0]);
      accg[1] = wmma_bf16(af, g1, accg[1]);
      accu[1] = wmma_bf16(af, u1, accu[1]);
      sched_fence();
    }
  }
  // Epilogue: SiLU gate, multiply, bf16 store. C/D layout: VGPR r -> row r + 8*hf.
#pragma unroll
  for (int ct = 0; ct < 2; ++ct) {
#pragma unroll
    for (int r = 0; r < 8; ++r) {
      float g = accg[ct][r], u = accu[ct][r];
      float a = (g / (1.f + __expf(-g))) * u;
      int row = m0 + wm * 16 + hf * 8 + r;
      int col = n0 + wn * 32 + ct * 16 + nl;
      act[(size_t)row * II + col] = (__bf16)a;
    }
  }
}

// ---------------------------------------------------------------------------
// GEMM2: out[t, h] (+)= w_all[t][e] * (act . w2[e][h, :]).  accum=0 overwrites
// (first expert) so poisoned d_out never needs a zero pass.
// ---------------------------------------------------------------------------
__global__ __launch_bounds__(256) void moe_gemm2(const bf16_t* __restrict__ act,
                                                 const float* __restrict__ w2,
                                                 const float* __restrict__ w_all,
                                                 float* __restrict__ out,
                                                 int e, int accum) {
  __shared__ __align__(16) bf16_t lA[BM * LDSS];
  __shared__ __align__(16) bf16_t lB[BN * LDSS];

  const int tid  = threadIdx.x;
  const int m0   = blockIdx.y * BM;
  const int n0   = blockIdx.x * BN;
  const int arow = tid >> 3;          // A (bf16 src): 32 rows/pass, 2 passes
  const int acol = (tid & 7) << 3;
  const int brow = tid >> 4;          // B (fp32 src): 16 rows/pass, 4 passes
  const int bcol = (tid & 15) << 2;
  const int wv = tid >> 5, lane = tid & 31;
  const int wm = wv >> 1, wn = wv & 1;
  const int nl = lane & 15, hf = lane >> 4;

  const bf16_t* aa = &act[(size_t)(m0 + arow) * II + acol];
  const float*  ba = &w2 [((size_t)e * HH + n0 + brow) * II + bcol];

  bf16x8 raq[2]; float4 rb[4];
#pragma unroll
  for (int p = 0; p < 2; ++p) raq[p] = *(const bf16x8*)(aa + (size_t)p * 32 * II);
#pragma unroll
  for (int p = 0; p < 4; ++p) rb[p]  = *(const float4*)(ba + (size_t)p * 16 * II);

  const v8f vz = {0.f,0.f,0.f,0.f,0.f,0.f,0.f,0.f};
  v8f acc[2] = {vz, vz};

  for (int kb = 0; kb < II; kb += BK) {
    __syncthreads();
#pragma unroll
    for (int p = 0; p < 2; ++p)
      *(bf16x8*)&lA[(p * 32 + arow) * LDSS + acol] = raq[p];
#pragma unroll
    for (int p = 0; p < 4; ++p)
      *(bf16x4*)&lB[(p * 16 + brow) * LDSS + bcol] = f4_to_bf4(rb[p]);
    __syncthreads();
    if (kb + BK < II) {
#pragma unroll
      for (int p = 0; p < 2; ++p) raq[p] = *(const bf16x8*)(aa + (size_t)p * 32 * II + kb + BK);
#pragma unroll
      for (int p = 0; p < 4; ++p) rb[p]  = *(const float4*)(ba + (size_t)p * 16 * II + kb + BK);
      if (kb + 2 * BK < II) __builtin_prefetch(ba + kb + 2 * BK, 0, 0);
    }
#pragma unroll
    for (int kk = 0; kk < BK; kk += 32) {
      const int b0 = (wn * 32 + nl) * LDSS + hf * 16 + kk;
      v16bf af  = ld_fragA(&lA[(wm * 16 + nl) * LDSS + hf * 16 + kk]);
      v16bf b0f = ld_fragB(&lB[b0]);
      v16bf b1f = ld_fragB(&lB[b0 + 16 * LDSS]);
      sched_fence();
      acc[0] = wmma_bf16(af, b0f, acc[0]);
      acc[1] = wmma_bf16(af, b1f, acc[1]);
      sched_fence();
    }
  }
#pragma unroll
  for (int ct = 0; ct < 2; ++ct) {
#pragma unroll
    for (int r = 0; r < 8; ++r) {
      int row = m0 + wm * 16 + hf * 8 + r;
      int col = n0 + wn * 32 + ct * 16 + nl;
      float wgt = w_all[row * EE + e];   // broadcast across the 16 lanes of a half-wave
      size_t o = (size_t)row * HH + col;
      float v = wgt * acc[ct][r];
      if (accum) out[o] = out[o] + v; else out[o] = v;
    }
  }
}

// ---------------------------------------------------------------------------
extern "C" void kernel_launch(void* const* d_in, const int* in_sizes, int n_in,
                              void* d_out, int out_size, void* d_ws, size_t ws_size,
                              hipStream_t stream) {
  const float* x      = (const float*)d_in[0];   // [T, H]
  const float* logits = (const float*)d_in[1];   // [T, E]
  const float* w13    = (const float*)d_in[2];   // [E, 2I, H]
  const float* w2     = (const float*)d_in[3];   // [E, H, I]
  (void)in_sizes; (void)n_in; (void)out_size; (void)ws_size;  // top_k fixed at 2

  float* out = (float*)d_out;                                  // [T, H]
  float* w_all = (float*)d_ws;                                 // T*E f32
  bf16_t* act = (bf16_t*)((char*)d_ws + (size_t)TT * EE * sizeof(float)); // T*I bf16

  moe_router<<<dim3(TT / 256), dim3(256), 0, stream>>>(logits, w_all);
  for (int e = 0; e < EE; ++e) {
    moe_gemm1<<<dim3(II / BN, TT / BM), dim3(256), 0, stream>>>(x, w13, act, e);
    moe_gemm2<<<dim3(HH / BN, TT / BM), dim3(256), 0, stream>>>(act, w2, w_all, out, e, e != 0);
  }
}
